// LiquidCausalGraph_88038239633733
// MI455X (gfx1250) — matrix operand order repaired and checked by reading, not compiled, not run on hardware
//
#include <hip/hip_runtime.h>
#include <hip/hip_bf16.h>
#include <stdint.h>

#define N_NODES 64
#define D_FEAT  64
#define B_BATCH 64
#define T_STEPS 32
#define DT_C    0.05f
#define SUBSTEPS 4

typedef __attribute__((ext_vector_type(16))) __bf16 v16bf;
typedef __attribute__((ext_vector_type(8)))  float  v8f;

__device__ __forceinline__ float sigmoidf_(float x){ return 1.0f/(1.0f + __expf(-x)); }

// ---- WMMA operand k order used in this file (contiguous, same for A and B):
//   A (16x32 bf16): k = e + (lane>=16 ? 16:0)
//   B (32x16 bf16): k = e + (lane>=16 ? 16:0)
// We store operands pre-swizzled as [ktile][tile][lane][e] so each lane's
// fragment is 32 contiguous bytes (2x ds_load_b128 instead of 16x ds_load_u16).

// swizzled index for an A-matrix element (row m, col k), tiles of 16x32, 4 row-tiles max
__device__ __forceinline__ int a_sw_idx(int m, int k, int nRowTiles){
  int ti = m >> 4, lanelow = m & 15;
  int kc = k >> 5, kk = k & 31;
  int lhalf = kk >> 4;
  int e = kk & 15;
  return ((kc*nRowTiles + ti)*32 + lhalf*16 + lanelow)*16 + e;
}
// swizzled index for a B-matrix element (row k, col d), tiles of 32x16, 4 col-tiles
__device__ __forceinline__ int b_sw_idx(int k, int d){
  int kc = k >> 5, kk = k & 31;
  int lhalf = kk >> 4;
  int e = kk & 15;
  int td = d >> 4, lanelow = d & 15;
  return ((kc*4 + td)*32 + lhalf*16 + lanelow)*16 + e;
}

// ---------------------------------------------------------------- utilities
__global__ __launch_bounds__(256) void f32_to_bf16_kernel(const float* __restrict__ src,
                                                          __bf16* __restrict__ dst, long n){
  long i = (long)blockIdx.x*blockDim.x + threadIdx.x;
  long stride = (long)gridDim.x*blockDim.x;
  for (; i < n; i += stride) dst[i] = (__bf16)src[i];
}

// per-node weights (node,128,64) -> bf16, pre-swizzled to B-fragment order per node
__global__ __launch_bounds__(256) void convert_w_swizzle_kernel(const float* __restrict__ src,
                                                                __bf16* __restrict__ dst){
  long i = (long)blockIdx.x*blockDim.x + threadIdx.x;
  long stride = (long)gridDim.x*blockDim.x;
  for (; i < 524288L; i += stride){
    int node = (int)(i >> 13);
    int rem  = (int)(i & 8191);
    int k = rem >> 6, d = rem & 63;
    dst[(long)node*8192 + b_sw_idx(k, d)] = (__bf16)src[i];
  }
}

__global__ __launch_bounds__(256) void zero_kernel(float* __restrict__ p, int n){
  int i = blockIdx.x*256 + threadIdx.x;
  if (i < n) p[i] = 0.0f;
}

// ---------------------------------------------------------------- A matrix
__global__ __launch_bounds__(256) void compute_A_kernel(const float* __restrict__ W,
    __bf16* __restrict__ A_bf, float* __restrict__ invtw, float* __restrict__ haspar,
    float* __restrict__ comb){
  __shared__ float A_s[4096];
  int tid = threadIdx.x;
  for (int idx = tid; idx < 4096; idx += 256){
    int i = idx >> 6, j = idx & 63;
    float w = W[idx];
    float a = sigmoidf_(w);
    if (i == j) a = 0.0f;
    if (a <= 0.01f) a = 0.0f;      // hard threshold
    A_s[idx] = a;
    A_bf[idx] = (__bf16)a;
    comb[idx] = w;                 // stage W flat -> graph_dyn input low half
  }
  __syncthreads();
  if (tid < 64){
    float tw = 0.0f;
    for (int j = 0; j < 64; ++j) tw += A_s[tid*64 + j];
    haspar[tid] = (tw > 1e-8f) ? 1.0f : 0.0f;
    invtw[tid]  = 1.0f / fmaxf(tw, 1e-8f);
  }
}

// ---------------------------------------------------------------- aggregate (A @ comb) per batch, WMMA bf16
// writes cat pre-swizzled (A-fragment order for node_ode): cat_sw[node][...] per 64x128 tile
__global__ __launch_bounds__(256) void aggregate_kernel(
    const float* __restrict__ emb, int t,
    const float* __restrict__ s_in,
    const __bf16* __restrict__ A_bf,
    const float* __restrict__ invtw, const float* __restrict__ haspar,
    __bf16* __restrict__ cat_sw){
  __shared__ __attribute__((aligned(32))) __bf16 A_s[4096];     // swizzled A-frag order
  __shared__ __attribute__((aligned(32))) __bf16 comb_s[4096];  // swizzled B-frag order
  int b = blockIdx.x;
  int tid = threadIdx.x;
  int lane = tid & 31;
  int wave = tid >> 5;
  long ebase = ((long)b*T_STEPS + t)*4096;
  for (int idx = tid; idx < 4096; idx += 256){
    int r = idx >> 6, c = idx & 63;
    A_s[a_sw_idx(r, c, 4)] = A_bf[idx];                   // (m=r, k=c)
    float cv = 0.5f*emb[ebase + idx] + 0.5f*s_in[(long)b*4096 + idx];
    comb_s[b_sw_idx(r, c)] = (__bf16)cv;                  // (k=r, d=c)
  }
  __syncthreads();
  for (int it = 0; it < 2; ++it){
    int tile = wave*2 + it;          // 16 tiles over 4x4 grid
    int ti = tile >> 2;              // output node tile
    int tn = tile & 3;               // feature tile
    v8f acc = {};
#pragma unroll
    for (int kc = 0; kc < 2; ++kc){
      v16bf af  = *reinterpret_cast<const v16bf*>(&A_s[((kc*4 + ti)*32 + lane)*16]);
      v16bf bfv = *reinterpret_cast<const v16bf*>(&comb_s[((kc*4 + tn)*32 + lane)*16]);
      acc = __builtin_amdgcn_wmma_f32_16x16x32_bf16(false, af, false, bfv,
                                                    (short)0, acc, false, false);
    }
#pragma unroll
    for (int r = 0; r < 8; ++r){
      int i_node = ti*16 + r + ((lane >= 16) ? 8 : 0);
      int d = tn*16 + (lane & 15);
      float p  = acc[r]*invtw[i_node];
      float e0 = emb[ebase + i_node*64 + d];
      float par = (haspar[i_node] > 0.5f) ? p : (e0*0.1f);
      // cat row = batch b, col k = d (parents in low half), A-fragment order per node
      cat_sw[(long)i_node*8192 + a_sw_idx(b, d, 4)] = (__bf16)par;
    }
  }
  // high half of cat (k = 64 + d) = stage states, same swizzled order
  for (int idx = tid; idx < 4096; idx += 256){
    int n = idx >> 6, d = idx & 63;
    cat_sw[(long)n*8192 + a_sw_idx(b, 64 + d, 4)] = (__bf16)s_in[(long)b*4096 + idx];
  }
}

// ---------------------------------------------------------------- node ODE: 3 per-node GEMMs + fused epilogue
__global__ __launch_bounds__(512) void node_ode_kernel(
    const __bf16* __restrict__ cat_sw,
    const __bf16* __restrict__ tau_w, const __bf16* __restrict__ f_w,
    const __bf16* __restrict__ gate_w,
    const float* __restrict__ tau_b, const float* __restrict__ f_b,
    const float* __restrict__ gate_b, const float* __restrict__ node_bias,
    const float* __restrict__ s_in, float* __restrict__ kx){
  __shared__ __attribute__((aligned(32))) __bf16 cat_s[8192];   // pre-swizzled A tiles
  __shared__ __attribute__((aligned(32))) __bf16 w_s[8192];     // pre-swizzled B tiles
  int node = blockIdx.x;
  int tid = threadIdx.x;
  int lane = tid & 31;
  int wave = tid >> 5;            // 16 waves, one 16x16 output tile each
  // contiguous vectorized stage of the node's cat tile
  {
    const uint4* src = reinterpret_cast<const uint4*>(cat_sw + (long)node*8192);
    uint4* dst = reinterpret_cast<uint4*>(cat_s);
    for (int idx = tid; idx < 1024; idx += 512) dst[idx] = src[idx];
  }
  __syncthreads();
  int ti = wave >> 2;             // batch tile
  int td = wave & 3;              // out-feature tile
  // cache A-fragments for all 4 K-chunks (shared across the 3 weight matrices)
  v16bf afc[4];
#pragma unroll
  for (int kc = 0; kc < 4; ++kc)
    afc[kc] = *reinterpret_cast<const v16bf*>(&cat_s[((kc*4 + ti)*32 + lane)*16]);
  long wbase = (long)node*8192;
  const __bf16* wlist[3] = { tau_w + wbase, f_w + wbase, gate_w + wbase };
  v8f accs[3] = {};
#pragma unroll
  for (int m = 0; m < 3; ++m){
    __syncthreads();
    {
      const uint4* src = reinterpret_cast<const uint4*>(wlist[m]);
      uint4* dst = reinterpret_cast<uint4*>(w_s);
      for (int idx = tid; idx < 1024; idx += 512) dst[idx] = src[idx];
    }
    __syncthreads();
#pragma unroll
    for (int kc = 0; kc < 4; ++kc){
      v16bf bfv = *reinterpret_cast<const v16bf*>(&w_s[((kc*4 + td)*32 + lane)*16]);
      accs[m] = __builtin_amdgcn_wmma_f32_16x16x32_bf16(false, afc[kc], false, bfv,
                                                        (short)0, accs[m], false, false);
    }
  }
#pragma unroll
  for (int r = 0; r < 8; ++r){
    int bb = ti*16 + r + ((lane >= 16) ? 8 : 0);
    int d  = td*16 + (lane & 15);
    int gi = node*64 + d;
    float tau = sigmoidf_(accs[0][r] + tau_b[gi]) + 0.1f;
    float fv  = tanhf(accs[1][r] + f_b[gi]) + node_bias[gi];
    float g   = sigmoidf_(accs[2][r] + gate_b[gi]);
    long sidx = (long)bb*4096 + gi;
    float s = s_in[sidx];
    kx[sidx] = g * (fv - s) / tau;
  }
}

// ---------------------------------------------------------------- graph_dyn: bf16 MV chain (L2-resident weights)
__global__ __launch_bounds__(256) void mean_kernel(const float* __restrict__ s_in,
                                                   float* __restrict__ sm){
  int i = blockIdx.x*256 + threadIdx.x;     // 0..4095
  float acc = 0.0f;
  for (int b = 0; b < 64; ++b) acc += s_in[(long)b*4096 + i];
  sm[i] = acc * (1.0f/64.0f);
}

__global__ __launch_bounds__(256) void mv_partial_kernel(const __bf16* __restrict__ w,
    const float* __restrict__ x, float* __restrict__ partial, int chunk){
  int j  = blockIdx.x*256 + threadIdx.x;    // output column (coalesced)
  int k0 = blockIdx.y * chunk;              // split-K slice
  const __bf16* wp = w + (long)k0*4096 + j;
  float acc = 0.0f;
  for (int i = 0; i < chunk; ++i){
    if ((i & 127) == 0) __builtin_prefetch(wp + (long)128*4096, 0, 1);
    acc += x[k0 + i] * (float)(*wp);
    wp += 4096;
  }
  partial[blockIdx.y*4096 + j] = acc;
}

__global__ __launch_bounds__(256) void mv_reduce_kernel(const float* __restrict__ partial,
    int splitK, const float* __restrict__ bias, float* __restrict__ out, int mode,
    const float* __restrict__ damping){
  int j = blockIdx.x*256 + threadIdx.x;
  float acc = bias[j];
  for (int s = 0; s < splitK; ++s) acc += partial[s*4096 + j];
  if (mode == 0){
    out[j] = tanhf(acc);
  } else {
    int i = j >> 6, jj = j & 63;
    float m  = (i == jj) ? 0.0f : 1.0f;
    float sd = sigmoidf_(damping[0]);
    out[j] = acc * m * sd * 0.1f;
  }
}

// ---------------------------------------------------------------- RK4 combination
__global__ __launch_bounds__(256) void advance_kernel(const float* __restrict__ base,
    const float* __restrict__ k, float* __restrict__ kacc, float* __restrict__ staged,
    float coef, float a, int assign, int n){
  int i = blockIdx.x*256 + threadIdx.x;
  if (i >= n) return;
  float kv = k[i];
  kacc[i]  = (assign ? 0.0f : kacc[i]) + coef*kv;
  staged[i] = base[i] + a*kv;
}

__global__ __launch_bounds__(256) void final_state_kernel(float* __restrict__ states,
    const float* __restrict__ kacc, const float* __restrict__ kx,
    float* __restrict__ traj, int t, int write_traj){
  int i = blockIdx.x*256 + threadIdx.x;     // < 262144
  float v = states[i] + (DT_C/6.0f)*(kacc[i] + kx[i]);
  v = fminf(5.0f, fmaxf(-5.0f, v));
  states[i] = v;
  if (write_traj){
    int b = i >> 12, r = i & 4095;
    traj[((long)b*T_STEPS + t)*4096 + r] = v;
  }
}

// W update + DAG projection: trace(expm(A∘A)) via truncated Taylor series in LDS
__global__ __launch_bounds__(256) void project_W_kernel(float* __restrict__ W,
    const float* __restrict__ kWacc, const float* __restrict__ kW4){
  __shared__ float M_s[4096];
  __shared__ float Pa[4096];
  __shared__ float Pb[4096];
  __shared__ float h_s;
  int tid = threadIdx.x;
  float wn[16];
  for (int q = 0; q < 16; ++q){
    int idx = tid + q*256;
    float w = W[idx] + (DT_C/6.0f)*(kWacc[idx] + kW4[idx]);
    wn[q] = w;
    int i = idx >> 6, j = idx & 63;
    float a = sigmoidf_(w);
    if (i == j) a = 0.0f;
    if (a <= 0.01f) a = 0.0f;
    float m = a*a;
    M_s[idx] = m;
    Pa[idx]  = m;
  }
  __syncthreads();
  if (tid == 0){
    float tr = 0.0f;
    for (int i = 0; i < 64; ++i) tr += M_s[i*64 + i];
    h_s = tr;
  }
  __syncthreads();
  float* cur = Pa; float* nxt = Pb;
  for (int k = 2; k <= 12; ++k){
    float inv = 1.0f/(float)k;
    for (int q = 0; q < 16; ++q){
      int idx = tid + q*256;
      int i = idx >> 6, j = idx & 63;
      float acc = 0.0f;
      for (int l = 0; l < 64; ++l) acc += cur[i*64 + l]*M_s[l*64 + j];
      nxt[idx] = acc*inv;
    }
    __syncthreads();
    if (tid == 0){
      float tr = 0.0f;
      for (int i = 0; i < 64; ++i) tr += nxt[i*64 + i];
      h_s += tr;
    }
    __syncthreads();
    float* tp = cur; cur = nxt; nxt = tp;
  }
  float h = h_s;                       // = trace(expm(M)) - N
  for (int q = 0; q < 16; ++q){
    int idx = tid + q*256;
    float w = wn[q];
    if (h > 0.1f){
      float sgn = (w > 0.0f) ? 1.0f : ((w < 0.0f) ? -1.0f : 0.0f);
      w = w - 0.05f*sgn*(h/64.0f);
    }
    W[idx] = w;
  }
}

// ---------------------------------------------------------------- host orchestration
extern "C" void kernel_launch(void* const* d_in, const int* in_sizes, int n_in,
                              void* d_out, int out_size, void* d_ws, size_t ws_size,
                              hipStream_t stream){
  const float* emb       = (const float*)d_in[0];
  const float* W_init    = (const float*)d_in[1];
  const float* tau_w     = (const float*)d_in[2];
  const float* tau_b     = (const float*)d_in[3];
  const float* f_w       = (const float*)d_in[4];
  const float* f_b       = (const float*)d_in[5];
  const float* gate_w    = (const float*)d_in[6];
  const float* gate_b    = (const float*)d_in[7];
  const float* node_bias = (const float*)d_in[8];
  const float* enc_w     = (const float*)d_in[9];
  const float* enc_b     = (const float*)d_in[10];
  const float* dyn1_w    = (const float*)d_in[11];
  const float* dyn1_b    = (const float*)d_in[12];
  const float* dyn2_w    = (const float*)d_in[13];
  const float* dyn2_b    = (const float*)d_in[14];
  const float* damping   = (const float*)d_in[15];
  (void)in_sizes; (void)n_in; (void)out_size; (void)ws_size;

  char* ws = (char*)d_ws;
  size_t off = 0;
  auto alloc = [&](size_t bytes)->char*{
    char* p = ws + off; off += (bytes + 255) & ~(size_t)255; return p;
  };
  __bf16* bf_enc  = (__bf16*)alloc(16777216UL*2);
  __bf16* bf_dyn1 = (__bf16*)alloc(33554432UL*2);
  __bf16* bf_dyn2 = (__bf16*)alloc(16777216UL*2);
  __bf16* bf_tau  = (__bf16*)alloc(524288UL*2);   // swizzled B-fragment order
  __bf16* bf_f    = (__bf16*)alloc(524288UL*2);
  __bf16* bf_gate = (__bf16*)alloc(524288UL*2);
  __bf16* bf_A    = (__bf16*)alloc(4096UL*2);
  __bf16* cat_sw  = (__bf16*)alloc(524288UL*2);   // swizzled A-fragment order per node
  float* states  = (float*)alloc(262144UL*4);
  float* s_stage = (float*)alloc(262144UL*4);
  float* kx      = (float*)alloc(262144UL*4);
  float* kacc    = (float*)alloc(262144UL*4);
  float* Wbuf    = (float*)alloc(4096UL*4);
  float* W_stage = (float*)alloc(4096UL*4);
  float* kW      = (float*)alloc(4096UL*4);
  float* kWacc   = (float*)alloc(4096UL*4);
  float* invtw   = (float*)alloc(64UL*4);
  float* haspar  = (float*)alloc(64UL*4);
  float* sm      = (float*)alloc(4096UL*4);
  float* comb    = (float*)alloc(8192UL*4);   // [W_stage flat | summ]
  float* hvec    = (float*)alloc(4096UL*4);
  float* partial = (float*)alloc(16UL*4096UL*4);

  // one-time bf16 conversion: graph_dyn weights (128 MB -> fits in 192 MB L2),
  // node weights pre-swizzled into WMMA B-fragment order
  f32_to_bf16_kernel<<<2048,256,0,stream>>>(enc_w,  bf_enc,  16777216L);
  f32_to_bf16_kernel<<<2048,256,0,stream>>>(dyn1_w, bf_dyn1, 33554432L);
  f32_to_bf16_kernel<<<2048,256,0,stream>>>(dyn2_w, bf_dyn2, 16777216L);
  convert_w_swizzle_kernel<<<512,256,0,stream>>>(tau_w,  bf_tau);
  convert_w_swizzle_kernel<<<512,256,0,stream>>>(f_w,    bf_f);
  convert_w_swizzle_kernel<<<512,256,0,stream>>>(gate_w, bf_gate);

  zero_kernel<<<1024,256,0,stream>>>(states, 262144);
  hipMemcpyAsync(Wbuf, W_init, 4096*sizeof(float), hipMemcpyDeviceToDevice, stream);

  auto run_stage = [&](const float* sin, const float* win, int t){
    compute_A_kernel<<<1,256,0,stream>>>(win, bf_A, invtw, haspar, comb);
    aggregate_kernel<<<64,256,0,stream>>>(emb, t, sin, bf_A, invtw, haspar, cat_sw);
    node_ode_kernel<<<64,512,0,stream>>>(cat_sw, bf_tau, bf_f, bf_gate,
                                         tau_b, f_b, gate_b, node_bias, sin, kx);
    mean_kernel<<<16,256,0,stream>>>(sin, sm);
    mv_partial_kernel<<<dim3(16,8),256,0,stream>>>(bf_enc, sm, partial, 512);
    mv_reduce_kernel<<<16,256,0,stream>>>(partial, 8, enc_b, comb + 4096, 0, damping);
    mv_partial_kernel<<<dim3(16,16),256,0,stream>>>(bf_dyn1, comb, partial, 512);
    mv_reduce_kernel<<<16,256,0,stream>>>(partial, 16, dyn1_b, hvec, 0, damping);
    mv_partial_kernel<<<dim3(16,8),256,0,stream>>>(bf_dyn2, hvec, partial, 512);
    mv_reduce_kernel<<<16,256,0,stream>>>(partial, 8, dyn2_b, kW, 1, damping);
  };

  for (int t = 0; t < T_STEPS; ++t){
    for (int sub = 0; sub < SUBSTEPS; ++sub){
      // stage 1
      run_stage(states, Wbuf, t);
      advance_kernel<<<1024,256,0,stream>>>(states, kx, kacc, s_stage, 1.0f, 0.5f*DT_C, 1, 262144);
      advance_kernel<<<16,256,0,stream>>>(Wbuf, kW, kWacc, W_stage, 1.0f, 0.5f*DT_C, 1, 4096);
      // stage 2
      run_stage(s_stage, W_stage, t);
      advance_kernel<<<1024,256,0,stream>>>(states, kx, kacc, s_stage, 2.0f, 0.5f*DT_C, 0, 262144);
      advance_kernel<<<16,256,0,stream>>>(Wbuf, kW, kWacc, W_stage, 2.0f, 0.5f*DT_C, 0, 4096);
      // stage 3
      run_stage(s_stage, W_stage, t);
      advance_kernel<<<1024,256,0,stream>>>(states, kx, kacc, s_stage, 2.0f, DT_C, 0, 262144);
      advance_kernel<<<16,256,0,stream>>>(Wbuf, kW, kWacc, W_stage, 2.0f, DT_C, 0, 4096);
      // stage 4 + combine + project
      run_stage(s_stage, W_stage, t);
      final_state_kernel<<<1024,256,0,stream>>>(states, kacc, kx, (float*)d_out, t,
                                                (sub == SUBSTEPS-1) ? 1 : 0);
      project_W_kernel<<<1,256,0,stream>>>(Wbuf, kWacc, kW);
    }
  }
  // W_final appended after trajectories
  hipMemcpyAsync((float*)d_out + 8388608L, Wbuf, 4096*sizeof(float),
                 hipMemcpyDeviceToDevice, stream);
}
